// MultiHeadAttention_84825604096278
// MI455X (gfx1250) — matrix operand order repaired and checked by reading, not compiled, not run on hardware
//
#include <hip/hip_runtime.h>
#include <hip/hip_bf16.h>

typedef __bf16 bf16_t;
typedef __attribute__((ext_vector_type(16))) __bf16 v16bf;
typedef __attribute__((ext_vector_type(8)))  __bf16 v8bf;
typedef __attribute__((ext_vector_type(4)))  __bf16 v4bf;
typedef __attribute__((ext_vector_type(8)))  float  v8f;
typedef __attribute__((ext_vector_type(4)))  float  v4f;

#define B_     2
#define S_     2048
#define DIN_   1024
#define DM_    1024
#define H_     16
#define DEPTH_ 64
#define NTOK_  (B_ * S_)

// Async 16B global->LDS copy (CDNA5 GLOBAL_LOAD_ASYNC_TO_LDS_B128, GVS mode).
// ldsPtr: destination in LDS; gBase: uniform global base (SGPR pair); gOff: per-lane byte offset.
#define ASYNC_CP16(ldsPtr, gBase, gOff)                                            \
    asm volatile("global_load_async_to_lds_b128 %0, %1, %2"                        \
                 :: "v"((unsigned)(uintptr_t)(ldsPtr)), "v"((unsigned)(gOff)),     \
                    "s"(gBase) : "memory")
#define ASYNC_WAIT() asm volatile("s_wait_asynccnt 0x0" ::: "memory")

// ---------------- WMMA fragment helpers (CDNA5 16x16x32 bf16 layouts) ----------------

// A 16x32 bf16: lanes 0-15 -> M=lane, K in [0,8)U[16,24); lanes 16-31 -> M=lane-16, K in [8,16)U[24,32).
__device__ __forceinline__ v16bf ld_frag_a(const bf16_t* lds, int stride, int m0, int k0, int lane) {
    const int r = m0 + (lane & 15);
    const int g = (lane & 16) ? 8 : 0;
    const bf16_t* p = lds + r * stride + k0 + g;
    v8bf lo = *(const v8bf*)(p);
    v8bf hi = *(const v8bf*)(p + 16);
    v16bf f;
#pragma unroll
    for (int i = 0; i < 8; ++i) { f[i] = lo[i]; f[i + 8] = hi[i]; }
    return f;
}

// B 32x16 bf16 from LDS stored transposed Bt[n][k]: lanes 0-15 -> N=lane, K in [0,8)U[16,24); lanes 16-31 -> K in [8,16)U[24,32).
__device__ __forceinline__ v16bf ld_frag_b(const bf16_t* ldsBt, int stride, int n0, int k0, int lane) {
    const int n = n0 + (lane & 15);
    const int k = k0 + ((lane & 16) ? 8 : 0);
    const bf16_t* p = ldsBt + n * stride + k;
    v8bf lo = *(const v8bf*)(p);
    v8bf hi = *(const v8bf*)(p + 16);
    v16bf f;
#pragma unroll
    for (int i = 0; i < 8; ++i) { f[i] = lo[i]; f[i + 8] = hi[i]; }
    return f;
}

__device__ __forceinline__ v8f wmma_bf16(v16bf a, v16bf b, v8f c) {
    return __builtin_amdgcn_wmma_f32_16x16x32_bf16(false, a, false, b, (short)0, c, false, false);
}

__device__ __forceinline__ void zero_acc(v8f* a, int n) {
    for (int i = 0; i < n; ++i)
#pragma unroll
        for (int e = 0; e < 8; ++e) a[i][e] = 0.0f;
}

#define PSA 40  // padded LDS stride (halfwords): 80B rows keep 16B chunk alignment
#define AST 72  // padded stride for 64-wide bf16 tiles (144B rows)

// ---------------- Kernel 1: fused QKV projection (f32 x f32 -> bf16 head-split) ----------------
// C[4096,1024] = X @ W + b, scattered to [B,H,S,64] bf16. 128x128 tile, K-tile 32, ping-pong LDS.

__global__ __launch_bounds__(256)
void qkv_proj_kernel(const float* __restrict__ xq, const float* __restrict__ xk, const float* __restrict__ xv,
                     const float* __restrict__ Wq, const float* __restrict__ bq,
                     const float* __restrict__ Wk, const float* __restrict__ bk,
                     const float* __restrict__ Wv, const float* __restrict__ bv,
                     bf16_t* __restrict__ Qh, bf16_t* __restrict__ Kh, bf16_t* __restrict__ Vh) {
    const float* X; const float* W; const float* bias; bf16_t* O;
    if (blockIdx.z == 0)      { X = xq; W = Wq; bias = bq; O = Qh; }
    else if (blockIdx.z == 1) { X = xk; W = Wk; bias = bk; O = Kh; }
    else                      { X = xv; W = Wv; bias = bv; O = Vh; }

    __shared__ alignas(16) bf16_t sA[2][128 * PSA];   // [m][k]
    __shared__ alignas(16) bf16_t sB[2][128 * PSA];   // transposed [n][k]

    const int tid = threadIdx.x;
    const int lane = tid & 31, wave = tid >> 5;
    const int wm = (wave & 3) * 32, wn = (wave >> 2) * 64;
    const int m_base = blockIdx.y * 128;
    const int n_base = blockIdx.x * 128;

    v8f acc[8];
    zero_acc(acc, 8);

    v4f ra[4], rb[4];
    auto load_stage = [&](int kc) {
#pragma unroll
        for (int i = 0; i < 4; ++i) {
            const int c = tid + i * 256;
            ra[i] = *(const v4f*)(X + (m_base + (c >> 3)) * DIN_ + kc + (c & 7) * 4);
            rb[i] = *(const v4f*)(W + (kc + (c >> 5)) * DM_ + n_base + (c & 31) * 4);
        }
    };
    auto store_stage = [&](int buf) {
#pragma unroll
        for (int i = 0; i < 4; ++i) {
            const int c = tid + i * 256;
            *(v4bf*)(sA[buf] + (c >> 3) * PSA + (c & 7) * 4) = __builtin_convertvector(ra[i], v4bf);
            const int br = c >> 5, bc = (c & 31) * 4;
#pragma unroll
            for (int e = 0; e < 4; ++e) sB[buf][(bc + e) * PSA + br] = (bf16_t)rb[i][e];
        }
    };

    load_stage(0);
    store_stage(0);
    __syncthreads();

    const int NK = DIN_ / 32;
    for (int ks = 0; ks < NK; ++ks) {
        const int cur = ks & 1;
        if (ks + 1 < NK) load_stage((ks + 1) * 32);
        v16bf af[2], bf[4];
#pragma unroll
        for (int m = 0; m < 2; ++m) af[m] = ld_frag_a(sA[cur], PSA, wm + m * 16, 0, lane);
#pragma unroll
        for (int n = 0; n < 4; ++n) bf[n] = ld_frag_b(sB[cur], PSA, wn + n * 16, 0, lane);
#pragma unroll
        for (int m = 0; m < 2; ++m)
#pragma unroll
            for (int n = 0; n < 4; ++n)
                acc[m * 4 + n] = wmma_bf16(af[m], bf[n], acc[m * 4 + n]);
        if (ks + 1 < NK) store_stage(cur ^ 1);
        __syncthreads();
    }

    const int rbase = m_base + wm + ((lane & 16) ? 8 : 0);
    const int cbase = n_base + wn + (lane & 15);
#pragma unroll
    for (int m = 0; m < 2; ++m)
#pragma unroll
        for (int n = 0; n < 4; ++n)
#pragma unroll
            for (int j = 0; j < 8; ++j) {
                const int row = rbase + m * 16 + j;
                const int col = cbase + n * 16;
                const float val = acc[m * 4 + n][j] + bias[col];
                const int b = row >> 11, s = row & (S_ - 1);
                const int h = col >> 6,  d = col & 63;
                O[(((size_t)(b * H_ + h) * S_ + s) * DEPTH_) + d] = (bf16_t)val;
            }
}

// ---------------- Kernel 2: logits = 0.125 * Qh @ Kh^T + mask*(-1e9) ----------------
// Pure bf16 tile copies -> async DMA to LDS (ASYNCcnt path). Kh rows are already B^T layout.

__global__ __launch_bounds__(256)
void logits_kernel(const bf16_t* __restrict__ Qh, const bf16_t* __restrict__ Kh,
                   const float* __restrict__ mask, float* __restrict__ attn) {
    const int bh = blockIdx.z;
    const int b = bh >> 4;
    const bf16_t* Q = Qh + (size_t)bh * S_ * DEPTH_;
    const bf16_t* K = Kh + (size_t)bh * S_ * DEPTH_;
    const int q_base = blockIdx.y * 128, k_base = blockIdx.x * 128;

    __shared__ alignas(16) bf16_t sQ[128 * AST];
    __shared__ alignas(16) bf16_t sK[128 * AST];

    const int tid = threadIdx.x, lane = tid & 31, wave = tid >> 5;
    const int wm = (wave & 3) * 32, wn = (wave >> 2) * 64;

    // 128x64 bf16 per tile = 1024 16B chunks; 4 per thread, DMA'd straight into LDS.
#pragma unroll
    for (int i = 0; i < 4; ++i) {
        const int c = tid + i * 256;
        const int r = c >> 3, c8 = (c & 7) * 8;
        ASYNC_CP16(sQ + r * AST + c8, Q, ((q_base + r) * DEPTH_ + c8) * 2);
        ASYNC_CP16(sK + r * AST + c8, K, ((k_base + r) * DEPTH_ + c8) * 2);
    }
    ASYNC_WAIT();
    __syncthreads();

    v8f acc[8];
    zero_acc(acc, 8);

#pragma unroll
    for (int k0 = 0; k0 < DEPTH_; k0 += 32) {
        v16bf af[2], bf[4];
#pragma unroll
        for (int m = 0; m < 2; ++m) af[m] = ld_frag_a(sQ, AST, wm + m * 16, k0, lane);
#pragma unroll
        for (int n = 0; n < 4; ++n) bf[n] = ld_frag_b(sK, AST, wn + n * 16, k0, lane);
#pragma unroll
        for (int m = 0; m < 2; ++m)
#pragma unroll
            for (int n = 0; n < 4; ++n)
                acc[m * 4 + n] = wmma_bf16(af[m], bf[n], acc[m * 4 + n]);
    }

    const int rbase = q_base + wm + ((lane & 16) ? 8 : 0);
    const int cbase = k_base + wn + (lane & 15);
#pragma unroll
    for (int m = 0; m < 2; ++m)
#pragma unroll
        for (int n = 0; n < 4; ++n) {
            const int kcol = cbase + n * 16;
            const float mterm = mask[b * S_ + kcol] * (-1e9f);
#pragma unroll
            for (int j = 0; j < 8; ++j) {
                const int qrow = rbase + m * 16 + j;
                attn[((size_t)bh * S_ + qrow) * S_ + kcol] = acc[m * 4 + n][j] * 0.125f + mterm;
            }
        }
}

// ---------------- Kernel 3: in-place row softmax over last dim (2048) ----------------

__global__ __launch_bounds__(256)
void softmax_kernel(float* __restrict__ attn) {
    float* p = attn + (size_t)blockIdx.x * S_;
    const int tid = threadIdx.x;

    v4f x0 = *(const v4f*)(p + tid * 4);
    v4f x1 = *(const v4f*)(p + (tid + 256) * 4);
    float m = -3.4e38f;
#pragma unroll
    for (int e = 0; e < 4; ++e) { m = fmaxf(m, x0[e]); m = fmaxf(m, x1[e]); }
#pragma unroll
    for (int off = 16; off; off >>= 1) m = fmaxf(m, __shfl_xor(m, off, 32));

    __shared__ float redm[8], reds[8];
    if ((tid & 31) == 0) redm[tid >> 5] = m;
    __syncthreads();
    float bm = redm[0];
#pragma unroll
    for (int i = 1; i < 8; ++i) bm = fmaxf(bm, redm[i]);

    float s = 0.0f;
#pragma unroll
    for (int e = 0; e < 4; ++e) {
        x0[e] = __expf(x0[e] - bm); s += x0[e];
        x1[e] = __expf(x1[e] - bm); s += x1[e];
    }
#pragma unroll
    for (int off = 16; off; off >>= 1) s += __shfl_xor(s, off, 32);
    if ((tid & 31) == 0) reds[tid >> 5] = s;
    __syncthreads();
    float tot = 0.0f;
#pragma unroll
    for (int i = 0; i < 8; ++i) tot += reds[i];
    const float inv = 1.0f / tot;

    *(v4f*)(p + tid * 4) = x0 * inv;
    *(v4f*)(p + (tid + 256) * 4) = x1 * inv;
}

// ---------------- Kernel 4: ctx = attn @ Vh (f32 attn -> bf16 on the fly) ----------------

__global__ __launch_bounds__(256)
void ctx_kernel(const float* __restrict__ attn, const bf16_t* __restrict__ Vh,
                bf16_t* __restrict__ ctx) {
    const int bh = blockIdx.z;
    const int b = bh >> 4, h = bh & 15;
    const float* Arow = attn + (size_t)bh * S_ * S_;
    const bf16_t* V = Vh + (size_t)bh * S_ * DEPTH_;
    const int q_base = blockIdx.y * 128;

    __shared__ alignas(16) bf16_t sA[2][128 * PSA];
    __shared__ alignas(16) bf16_t sBt[2][DEPTH_ * PSA];

    const int tid = threadIdx.x, lane = tid & 31, wave = tid >> 5;
    const int wm = (wave & 3) * 32, wn = (wave >> 2) * 32;

    v8f acc[4];
    zero_acc(acc, 4);

    v4f ra[4]; v8bf rv;
    auto load_stage = [&](int kc) {
#pragma unroll
        for (int i = 0; i < 4; ++i) {
            const int c = tid + i * 256;
            ra[i] = *(const v4f*)(Arow + (size_t)(q_base + (c >> 3)) * S_ + kc + (c & 7) * 4);
        }
        rv = *(const v8bf*)(V + (kc + (tid >> 3)) * DEPTH_ + (tid & 7) * 8);
    };
    auto store_stage = [&](int buf) {
#pragma unroll
        for (int i = 0; i < 4; ++i) {
            const int c = tid + i * 256;
            *(v4bf*)(sA[buf] + (c >> 3) * PSA + (c & 7) * 4) = __builtin_convertvector(ra[i], v4bf);
        }
        const int vr = tid >> 3, vc = (tid & 7) * 8;
#pragma unroll
        for (int e = 0; e < 8; ++e) sBt[buf][(vc + e) * PSA + vr] = rv[e];
    };

    load_stage(0);
    store_stage(0);
    __syncthreads();

    const int NK = S_ / 32;
    for (int ks = 0; ks < NK; ++ks) {
        const int cur = ks & 1;
        if (ks + 1 < NK) load_stage((ks + 1) * 32);
        v16bf af[2], bf[2];
#pragma unroll
        for (int m = 0; m < 2; ++m) af[m] = ld_frag_a(sA[cur], PSA, wm + m * 16, 0, lane);
#pragma unroll
        for (int n = 0; n < 2; ++n) bf[n] = ld_frag_b(sBt[cur], PSA, wn + n * 16, 0, lane);
#pragma unroll
        for (int m = 0; m < 2; ++m)
#pragma unroll
            for (int n = 0; n < 2; ++n)
                acc[m * 2 + n] = wmma_bf16(af[m], bf[n], acc[m * 2 + n]);
        if (ks + 1 < NK) store_stage(cur ^ 1);
        __syncthreads();
    }

    const int rbase = q_base + wm + ((lane & 16) ? 8 : 0);
    const int cbase = wn + (lane & 15);
#pragma unroll
    for (int m = 0; m < 2; ++m)
#pragma unroll
        for (int n = 0; n < 2; ++n)
#pragma unroll
            for (int j = 0; j < 8; ++j) {
                const int qrow = rbase + m * 16 + j;
                const int d = cbase + n * 16;
                ctx[((size_t)(b * S_ + qrow)) * DM_ + h * DEPTH_ + d] = (bf16_t)acc[m * 2 + n][j];
            }
}

// ---------------- Kernel 5: out = ctx @ Wo + bo ----------------
// A (ctx) is already bf16: tile copies go through async DMA into LDS; W converts via registers.

__global__ __launch_bounds__(256)
void out_proj_kernel(const bf16_t* __restrict__ Xc, const float* __restrict__ Wo,
                     const float* __restrict__ bo, float* __restrict__ out) {
    __shared__ alignas(16) bf16_t sA[2][128 * PSA];
    __shared__ alignas(16) bf16_t sB[2][128 * PSA];

    const int tid = threadIdx.x, lane = tid & 31, wave = tid >> 5;
    const int wm = (wave & 3) * 32, wn = (wave >> 2) * 64;
    const int m_base = blockIdx.y * 128;
    const int n_base = blockIdx.x * 128;

    v8f acc[8];
    zero_acc(acc, 8);

    v4f rb[4];
    auto issue_a_async = [&](int kc, int buf) {
#pragma unroll
        for (int i = 0; i < 2; ++i) {       // 128x32 bf16 = 512 16B chunks, 2 per thread
            const int c = tid + i * 256;
            const int r = c >> 2, c8 = (c & 3) * 8;
            ASYNC_CP16(sA[buf] + r * PSA + c8, Xc, ((m_base + r) * DM_ + kc + c8) * 2);
        }
    };
    auto load_b = [&](int kc) {
#pragma unroll
        for (int i = 0; i < 4; ++i) {
            const int c = tid + i * 256;
            rb[i] = *(const v4f*)(Wo + (kc + (c >> 5)) * DM_ + n_base + (c & 31) * 4);
        }
    };
    auto store_b = [&](int buf) {
#pragma unroll
        for (int i = 0; i < 4; ++i) {
            const int c = tid + i * 256;
            const int br = c >> 5, bc = (c & 31) * 4;
#pragma unroll
            for (int e = 0; e < 4; ++e) sB[buf][(bc + e) * PSA + br] = (bf16_t)rb[i][e];
        }
    };

    issue_a_async(0, 0);
    load_b(0);
    store_b(0);
    ASYNC_WAIT();
    __syncthreads();

    const int NK = DM_ / 32;
    for (int ks = 0; ks < NK; ++ks) {
        const int cur = ks & 1;
        if (ks + 1 < NK) { issue_a_async((ks + 1) * 32, cur ^ 1); load_b((ks + 1) * 32); }
        v16bf af[2], bf[4];
#pragma unroll
        for (int m = 0; m < 2; ++m) af[m] = ld_frag_a(sA[cur], PSA, wm + m * 16, 0, lane);
#pragma unroll
        for (int n = 0; n < 4; ++n) bf[n] = ld_frag_b(sB[cur], PSA, wn + n * 16, 0, lane);
#pragma unroll
        for (int m = 0; m < 2; ++m)
#pragma unroll
            for (int n = 0; n < 4; ++n)
                acc[m * 4 + n] = wmma_bf16(af[m], bf[n], acc[m * 4 + n]);
        if (ks + 1 < NK) store_b(cur ^ 1);
        ASYNC_WAIT();
        __syncthreads();
    }

    const int rbase = m_base + wm + ((lane & 16) ? 8 : 0);
    const int cbase = n_base + wn + (lane & 15);
#pragma unroll
    for (int m = 0; m < 2; ++m)
#pragma unroll
        for (int n = 0; n < 4; ++n)
#pragma unroll
            for (int j = 0; j < 8; ++j) {
                const int row = rbase + m * 16 + j;
                const int col = cbase + n * 16;
                out[(size_t)row * DM_ + col] = acc[m * 4 + n][j] + bo[col];
            }
}

// ---------------- launch ----------------

extern "C" void kernel_launch(void* const* d_in, const int* in_sizes, int n_in,
                              void* d_out, int out_size, void* d_ws, size_t ws_size,
                              hipStream_t stream) {
    const float* v    = (const float*)d_in[0];
    const float* k    = (const float*)d_in[1];
    const float* q    = (const float*)d_in[2];
    const float* mask = (const float*)d_in[3];
    const float* Wq   = (const float*)d_in[4];
    const float* bq   = (const float*)d_in[5];
    const float* Wk   = (const float*)d_in[6];
    const float* bk   = (const float*)d_in[7];
    const float* Wv   = (const float*)d_in[8];
    const float* bv   = (const float*)d_in[9];
    const float* Wo   = (const float*)d_in[10];
    const float* bo   = (const float*)d_in[11];

    float* out  = (float*)d_out;                       // [B,S,DM] f32
    float* attn = out + (size_t)B_ * S_ * DM_;         // [B,H,S,S] f32

    const size_t hsz = (size_t)B_ * H_ * S_ * DEPTH_;  // 4M elems
    bf16_t* Qh  = (bf16_t*)d_ws;
    bf16_t* Kh  = Qh + hsz;
    bf16_t* Vh  = Kh + hsz;
    bf16_t* ctx = Vh + hsz;                            // [B,S,DM] bf16

    qkv_proj_kernel<<<dim3(DM_ / 128, NTOK_ / 128, 3), 256, 0, stream>>>(
        q, k, v, Wq, bq, Wk, bk, Wv, bv, Qh, Kh, Vh);
    logits_kernel<<<dim3(S_ / 128, S_ / 128, B_ * H_), 256, 0, stream>>>(Qh, Kh, mask, attn);
    softmax_kernel<<<dim3(B_ * H_ * S_), 256, 0, stream>>>(attn);
    ctx_kernel<<<dim3(1, S_ / 128, B_ * H_), 256, 0, stream>>>(attn, Vh, ctx);
    out_proj_kernel<<<dim3(DM_ / 128, NTOK_ / 128), 256, 0, stream>>>(ctx, Wo, bo, out);
}
